// EncoderBlock_13151189860426
// MI455X (gfx1250) — matrix operand order repaired and checked by reading, not compiled, not run on hardware
//
#include <hip/hip_runtime.h>
#include <hip/hip_bf16.h>
#include <math.h>

// Problem constants (match reference)
#define B_    4
#define S_    1024
#define D_    1024
#define H_    16
#define DK_   64
#define DFF_  4096
#define MROWS (B_ * S_)   // 4096
#define EPS_  1e-9f

typedef unsigned short u16;
typedef __attribute__((ext_vector_type(16))) __bf16 v16bf;
typedef __attribute__((ext_vector_type(8)))  float  v8f;

union Frag16 { v16bf bf; uint4 q[2]; };   // 32 bytes: one 16-bit WMMA A/B fragment
union U4     { uint4 q; u16 s[8]; };

__device__ __forceinline__ u16 f2bf(float f) {
  // round-to-nearest-even fp32 -> bf16
  unsigned x = __float_as_uint(f);
  unsigned r = x + 0x7fffu + ((x >> 16) & 1u);
  return (u16)(r >> 16);
}

// -------------------------------------------------------------------------
// Weight convert + transpose: dst[n*K + k] = bf16(src[k*N + n])
// (so WMMA B fragments read two K-contiguous 16-byte chunks per lane)
// -------------------------------------------------------------------------
__global__ void wconv_kernel(const float* __restrict__ src, u16* __restrict__ dst,
                             int K, int N) {
  int idx = blockIdx.x * blockDim.x + threadIdx.x;
  if (idx >= K * N) return;
  int k = idx % K;
  int n = idx / K;
  dst[idx] = f2bf(src[(size_t)k * N + n]);
}

// -------------------------------------------------------------------------
// LayerNorm over last dim D_ (torch-style: unbiased std, eps added to std).
// One 256-thread block per row; optional fp32 and/or bf16 outputs.
// In-place (in == outF) is safe: each thread touches only its own elements.
// -------------------------------------------------------------------------
__global__ void ln_kernel(const float* __restrict__ in, float* __restrict__ outF,
                          u16* __restrict__ outB,
                          const float* __restrict__ alpha, const float* __restrict__ beta) {
  __shared__ float red[256];
  int row = blockIdx.x;
  int t   = threadIdx.x;
  const float* xr = in + (size_t)row * D_;

  float xv[4];
  float s = 0.f;
  #pragma unroll
  for (int i = 0; i < 4; ++i) { xv[i] = xr[t + i * 256]; s += xv[i]; }
  red[t] = s; __syncthreads();
  for (int off = 128; off; off >>= 1) { if (t < off) red[t] += red[t + off]; __syncthreads(); }
  float mean = red[0] * (1.f / D_);
  __syncthreads();

  float ss = 0.f;
  #pragma unroll
  for (int i = 0; i < 4; ++i) { float c = xv[i] - mean; ss += c * c; }
  red[t] = ss; __syncthreads();
  for (int off = 128; off; off >>= 1) { if (t < off) red[t] += red[t + off]; __syncthreads(); }
  float stdv = sqrtf(red[0] / (float)(D_ - 1));

  float inv = alpha[0] / (stdv + EPS_);
  float bb  = beta[0];
  #pragma unroll
  for (int i = 0; i < 4; ++i) {
    float y = (xv[i] - mean) * inv + bb;
    size_t o = (size_t)row * D_ + t + i * 256;
    if (outF) outF[o] = y;
    if (outB) outB[o] = f2bf(y);
  }
}

// -------------------------------------------------------------------------
// bf16 GEMM: C[M][N] = A[M][K] (bf16) x Wt[N][K] (bf16, pre-transposed)
//            + bias (opt) -> relu (opt) -> + residual (opt)
// 256 threads = 8 waves; wave tile 32(M) x 64(N) = 2x4 WMMA C frags.
// Block tile 128x128. M,N,K all multiples of 128/32 here -> no bounds checks.
// -------------------------------------------------------------------------
__global__ void __launch_bounds__(256)
gemm_bf16_kernel(const u16* __restrict__ A, const u16* __restrict__ Wt,
                 const float* __restrict__ bias, const float* __restrict__ res,
                 float* __restrict__ outF, u16* __restrict__ outB,
                 int M, int N, int K, int relu) {
  int lane  = threadIdx.x & 31;
  int wid   = threadIdx.x >> 5;        // 0..7
  int hi    = lane >> 4;               // half-wave
  int ln    = lane & 15;
  int mbase = blockIdx.y * 128 + (wid >> 1) * 32;
  int nbase = blockIdx.x * 128 + (wid & 1) * 64;
  (void)M;

  v8f zero = {0.f,0.f,0.f,0.f,0.f,0.f,0.f,0.f};
  v8f acc[2][4];
  #pragma unroll
  for (int i = 0; i < 2; ++i)
    #pragma unroll
    for (int j = 0; j < 4; ++j) acc[i][j] = zero;

  const u16* arow0 = A + (size_t)(mbase + ln) * K;
  const u16* arow1 = A + (size_t)(mbase + 16 + ln) * K;

  for (int k0 = 0; k0 < K; k0 += 32) {
    // A fragments: 16-bit A layout -> element e holds K = k0 + e + hi*8 + (e>=8)*8
    Frag16 a[2];
    a[0].q[0] = *(const uint4*)(arow0 + k0 + hi * 8);
    a[0].q[1] = *(const uint4*)(arow0 + k0 + 16 + hi * 8);
    a[1].q[0] = *(const uint4*)(arow1 + k0 + hi * 8);
    a[1].q[1] = *(const uint4*)(arow1 + k0 + 16 + hi * 8);

    __builtin_prefetch(arow0 + k0 + 128, 0, 1);   // global_prefetch_b8

    #pragma unroll
    for (int nf = 0; nf < 4; ++nf) {
      // B fragment: element e holds K = k0 + e + hi*8 + (e>=8)*8, col n = ln
      const u16* brow = Wt + (size_t)(nbase + nf * 16 + ln) * K + k0;
      Frag16 b;
      b.q[0] = *(const uint4*)(brow + hi * 8);
      b.q[1] = *(const uint4*)(brow + 16 + hi * 8);
      acc[0][nf] = __builtin_amdgcn_wmma_f32_16x16x32_bf16(
          false, a[0].bf, false, b.bf, (short)0, acc[0][nf], false, false);
      acc[1][nf] = __builtin_amdgcn_wmma_f32_16x16x32_bf16(
          false, a[1].bf, false, b.bf, (short)0, acc[1][nf], false, false);
    }
  }

  // Epilogue. C layout: VGPR v / lane -> row m = v + hi*8, col n = ln
  #pragma unroll
  for (int mf = 0; mf < 2; ++mf) {
    #pragma unroll
    for (int nf = 0; nf < 4; ++nf) {
      int n = nbase + nf * 16 + ln;
      float bn = bias ? bias[n] : 0.f;
      #pragma unroll
      for (int v = 0; v < 8; ++v) {
        int m = mbase + mf * 16 + v + hi * 8;
        float val = acc[mf][nf][v] + bn;
        if (relu) val = val > 0.f ? val : 0.f;
        size_t o = (size_t)m * N + n;
        if (res)  val += res[o];
        if (outF) outF[o] = val;
        if (outB) outB[o] = f2bf(val);
      }
    }
  }
}

// -------------------------------------------------------------------------
// Flash attention (q=k=v=p, shared projection "bug" preserved).
// Block = 128 threads (4 waves) per (b*H+h, 64-row q block); each wave does
// 16 q rows. KV tiles of 32 keys staged in LDS; online softmax in fp32.
// -------------------------------------------------------------------------
__global__ void __launch_bounds__(128)
attn_kernel(const u16* __restrict__ P, u16* __restrict__ Aout) {
  __shared__ u16 kv[32][64];          // key-tile row-major  [key][dk]
  __shared__ u16 vt[64][32];          // key-tile transposed [dk][key]
  __shared__ u16 plds[4][16][32];     // per-wave P staging (C layout -> A layout)

  int t    = threadIdx.x;
  int lane = t & 31, wid = t >> 5;
  int hi   = lane >> 4, ln = lane & 15;
  int bh   = blockIdx.x;
  int b    = bh / H_, h = bh % H_;
  int qbase = blockIdx.y * 64 + wid * 16;

  const u16* Pb = P + (size_t)b * S_ * D_ + (size_t)h * DK_;

  // Q fragments for this wave's 16 rows (2 frags cover DK=64)
  Frag16 qf[2];
  const u16* qrow = Pb + (size_t)(qbase + ln) * D_;
  #pragma unroll
  for (int f = 0; f < 2; ++f) {
    qf[f].q[0] = *(const uint4*)(qrow + f * 32 + hi * 8);
    qf[f].q[1] = *(const uint4*)(qrow + f * 32 + 16 + hi * 8);
  }

  v8f zero = {0.f,0.f,0.f,0.f,0.f,0.f,0.f,0.f};
  v8f acc[4];
  #pragma unroll
  for (int d = 0; d < 4; ++d) acc[d] = zero;
  float mrow[8], lrow[8];
  #pragma unroll
  for (int v = 0; v < 8; ++v) { mrow[v] = -__builtin_inff(); lrow[v] = 0.f; }

  int krow = t >> 2;              // 0..31 key within tile
  int c0   = (t & 3) * 16;        // dk chunk

  for (int kt = 0; kt < S_ / 32; ++kt) {
    // cooperative KV tile load: 128 threads x 16 bf16 = 32x64
    const u16* src = Pb + (size_t)(kt * 32 + krow) * D_ + c0;
    U4 u0, u1;
    u0.q = *(const uint4*)(src);
    u1.q = *(const uint4*)(src + 8);
    *(uint4*)&kv[krow][c0]     = u0.q;
    *(uint4*)&kv[krow][c0 + 8] = u1.q;
    #pragma unroll
    for (int j = 0; j < 8; ++j) {
      vt[c0 + j][krow]     = u0.s[j];
      vt[c0 + 8 + j][krow] = u1.s[j];
    }
    __syncthreads();

    // scores S = Q * K^T / 8  (two 16x16 C frags = 32 keys)
    v8f sm[2];
    #pragma unroll
    for (int sf = 0; sf < 2; ++sf) {
      v8f s = zero;
      #pragma unroll
      for (int ks = 0; ks < 2; ++ks) {
        Frag16 kb;  // B[kk][n] = K[n][kk]: row n of kv, kk = ks*32 + e + hi*8 + (e>=8)*8
        const u16* kr = &kv[sf * 16 + ln][ks * 32];
        kb.q[0] = *(const uint4*)(kr + hi * 8);
        kb.q[1] = *(const uint4*)(kr + 16 + hi * 8);
        s = __builtin_amdgcn_wmma_f32_16x16x32_bf16(
            false, qf[ks].bf, false, kb.bf, (short)0, s, false, false);
      }
      sm[sf] = s * 0.125f;   // 1/sqrt(DK)
    }

    // online softmax (row reductions across the 16-lane half-wave)
    float sc[8];
    #pragma unroll
    for (int v = 0; v < 8; ++v) {
      float tm = fmaxf(sm[0][v], sm[1][v]);
      #pragma unroll
      for (int msk = 8; msk; msk >>= 1) tm = fmaxf(tm, __shfl_xor(tm, msk, 32));
      float mn = fmaxf(mrow[v], tm);
      float fz = __expf(mrow[v] - mn);
      float p0 = __expf(sm[0][v] - mn);
      float p1 = __expf(sm[1][v] - mn);
      sm[0][v] = p0; sm[1][v] = p1;
      float rs = p0 + p1;
      #pragma unroll
      for (int msk = 8; msk; msk >>= 1) rs += __shfl_xor(rs, msk, 32);
      lrow[v] = lrow[v] * fz + rs;
      mrow[v] = mn;
      sc[v]   = fz;
    }
    #pragma unroll
    for (int d = 0; d < 4; ++d)
      #pragma unroll
      for (int v = 0; v < 8; ++v) acc[d][v] *= sc[v];

    // stage P (C layout f32) -> per-wave LDS as bf16
    #pragma unroll
    for (int sf = 0; sf < 2; ++sf)
      #pragma unroll
      for (int v = 0; v < 8; ++v)
        plds[wid][v + hi * 8][sf * 16 + ln] = f2bf(sm[sf][v]);
    __syncthreads();

    // O += P(16x32) * V(32x64): read P back in A layout, V from transposed tile
    Frag16 pa;
    const u16* pr = &plds[wid][ln][0];
    pa.q[0] = *(const uint4*)(pr + hi * 8);
    pa.q[1] = *(const uint4*)(pr + 16 + hi * 8);
    #pragma unroll
    for (int d = 0; d < 4; ++d) {
      Frag16 vb;  // B[kk][dim] = V[kk][dim]: vt row dim, kk = e + hi*8 + (e>=8)*8
      const u16* vr = &vt[d * 16 + ln][0];
      vb.q[0] = *(const uint4*)(vr + hi * 8);
      vb.q[1] = *(const uint4*)(vr + 16 + hi * 8);
      acc[d] = __builtin_amdgcn_wmma_f32_16x16x32_bf16(
          false, pa.bf, false, vb.bf, (short)0, acc[d], false, false);
    }
    __syncthreads();   // before next tile overwrites kv/vt/plds
  }

  // normalize and write bf16 output (feeds the wo GEMM)
  u16* orow = Aout + (size_t)b * S_ * D_ + (size_t)h * DK_;
  #pragma unroll
  for (int d = 0; d < 4; ++d) {
    #pragma unroll
    for (int v = 0; v < 8; ++v) {
      int q = qbase + v + hi * 8;
      float o = acc[d][v] / lrow[v];
      orow[(size_t)q * D_ + d * 16 + ln] = f2bf(o);
    }
  }
}

// -------------------------------------------------------------------------
// Host-side orchestration
// -------------------------------------------------------------------------
extern "C" void kernel_launch(void* const* d_in, const int* in_sizes, int n_in,
                              void* d_out, int out_size, void* d_ws, size_t ws_size,
                              hipStream_t stream) {
  (void)in_sizes; (void)n_in; (void)out_size; (void)ws_size;

  const float* x      = (const float*)d_in[0];
  const float* wq_w   = (const float*)d_in[1];
  const float* wq_b   = (const float*)d_in[2];
  const float* wo_w   = (const float*)d_in[3];
  const float* wo_b   = (const float*)d_in[4];
  const float* ff1_w  = (const float*)d_in[5];
  const float* ff1_b  = (const float*)d_in[6];
  const float* ff2_w  = (const float*)d_in[7];
  const float* ff2_b  = (const float*)d_in[8];
  const float* alpha1 = (const float*)d_in[9];
  const float* bias1  = (const float*)d_in[10];
  const float* alpha2 = (const float*)d_in[11];
  const float* bias2  = (const float*)d_in[12];
  const float* alpha3 = (const float*)d_in[13];
  const float* bias3  = (const float*)d_in[14];
  float* out = (float*)d_out;

  char* ws = (char*)d_ws;
  const size_t MB = 1024 * 1024;
  u16*   wq_t  = (u16*)  (ws + 0 * MB);   //  2 MB  [D][D]   bf16 transposed
  u16*   wo_t  = (u16*)  (ws + 2 * MB);   //  2 MB
  u16*   ff1_t = (u16*)  (ws + 4 * MB);   //  8 MB  [DFF][D]
  u16*   ff2_t = (u16*)  (ws + 12 * MB);  //  8 MB  [D][DFF]
  u16*   lnbuf = (u16*)  (ws + 20 * MB);  //  8 MB  ln1 then ln2 (bf16)
  u16*   p_bf  = (u16*)  (ws + 28 * MB);  //  8 MB  shared q/k/v projection
  u16*   at_bf = (u16*)  (ws + 36 * MB);  //  8 MB  attention output
  float* h1    = (float*)(ws + 44 * MB);  // 16 MB  residual stream
  u16*   f_bf  = (u16*)  (ws + 60 * MB);  // 32 MB  relu(ffn hidden)  -> 92 MB total

  // 1) weights -> bf16, transposed for B-fragment-friendly layout
  wconv_kernel<<<(D_ * D_) / 256, 256, 0, stream>>>(wq_w, wq_t, D_, D_);
  wconv_kernel<<<(D_ * D_) / 256, 256, 0, stream>>>(wo_w, wo_t, D_, D_);
  wconv_kernel<<<(D_ * DFF_) / 256, 256, 0, stream>>>(ff1_w, ff1_t, D_, DFF_);
  wconv_kernel<<<(D_ * DFF_) / 256, 256, 0, stream>>>(ff2_w, ff2_t, DFF_, D_);

  // 2) ln1(x) -> bf16
  ln_kernel<<<MROWS, 256, 0, stream>>>(x, nullptr, lnbuf, alpha1, bias1);

  // 3) p = ln1 @ wq + wq_b   (bf16; used for q, k AND v per the reference bug)
  gemm_bf16_kernel<<<dim3(D_ / 128, MROWS / 128), 256, 0, stream>>>(
      lnbuf, wq_t, wq_b, nullptr, nullptr, p_bf, MROWS, D_, D_, 0);

  // 4) attention
  attn_kernel<<<dim3(B_ * H_, S_ / 64), 128, 0, stream>>>(p_bf, at_bf);

  // 5) h1 = x + attn @ wo + wo_b   (fp32)
  gemm_bf16_kernel<<<dim3(D_ / 128, MROWS / 128), 256, 0, stream>>>(
      at_bf, wo_t, wo_b, x, h1, nullptr, MROWS, D_, D_, 0);

  // 6) ln2(h1) -> bf16
  ln_kernel<<<MROWS, 256, 0, stream>>>(h1, nullptr, lnbuf, alpha2, bias2);

  // 7) f = relu(ln2 @ ff1 + b1)   (bf16)
  gemm_bf16_kernel<<<dim3(DFF_ / 128, MROWS / 128), 256, 0, stream>>>(
      lnbuf, ff1_t, ff1_b, nullptr, nullptr, f_bf, MROWS, DFF_, D_, 1);

  // 8) h2 = h1 + f @ ff2 + b2 -> d_out (fp32)
  gemm_bf16_kernel<<<dim3(D_ / 128, MROWS / 128), 256, 0, stream>>>(
      f_bf, ff2_t, ff2_b, h1, out, nullptr, MROWS, D_, DFF_, 0);

  // 9) ln3 in-place on d_out
  ln_kernel<<<MROWS, 256, 0, stream>>>(out, out, nullptr, alpha3, bias3);
}
